// NeuralPointNet_28080496181704
// MI455X (gfx1250) — compile-verified, matrix operand order checked by reading, not run on hardware
//
#include <hip/hip_runtime.h>
#include <cstdint>
#include <cstddef>

// ---------------------------------------------------------------------------
// CDNA5 (gfx1250) wave32 WMMA types
// ---------------------------------------------------------------------------
typedef __attribute__((ext_vector_type(16))) _Float16 v16h;
typedef __attribute__((ext_vector_type(8)))  _Float16 v8h;
typedef __attribute__((ext_vector_type(8)))  float    v8f;

#define DEVINL __device__ __forceinline__

// Load one 16x32 f16 fragment (A or B operand) from a row-major matrix with
// row stride S=Kp halfs. wave32 layout (CDNA5 ISA, 16-bit A 16x32):
//   lanes 0-15 : row = lane,   K in {0..7, 16..23}
//   lanes 16-31: row = lane-16,K in {8..15, 24..31}
// Per lane that is two contiguous 8-half (16B) chunks -> two b128 loads.
DEVINL v16h ldfrag(const _Float16* __restrict__ p) {
  v8h lo = *(const v8h*)(p);
  v8h hi = *(const v8h*)(p + 16);
  v16h r;
#pragma unroll
  for (int i = 0; i < 8; ++i) { r[i] = lo[i]; r[i + 8] = hi[i]; }
  return r;
}

// Store one 16x16 f32 accumulator tile with bias + ReLU.
// C/D layout: lanes 0-15 -> M=0..7 (vgpr r => M=r), lanes 16-31 -> M=8..15;
// N = lane & 15.
DEVINL void store_tile(const v8f& acc, _Float16* __restrict__ Y,
                       float* __restrict__ Yf, const float* __restrict__ bias,
                       int mBase, int nBase, int O, int Op, int lane) {
  int n  = nBase + (lane & 15);
  int m0 = mBase + ((lane >> 4) << 3);
  float bv = (n < O) ? bias[n] : 0.0f;
#pragma unroll
  for (int r = 0; r < 8; ++r) {
    float v = acc[r] + bv;
    v = v > 0.0f ? v : 0.0f;                    // all MLP layers here are ReLU
    size_t m = (size_t)(m0 + r);
    Y[m * (size_t)Op + n] = (_Float16)v;
    if (Yf != nullptr && n < O) Yf[m * (size_t)O + n] = v;
  }
}

// ---------------------------------------------------------------------------
// GEMM: Y(M x Op, f16, ReLU) [+ optional f32 copy] = X(M x Kp, f16) * Wh^T + b
// Wh is the f16, zero-padded (Op x Kp) weight matrix (row o = output channel).
// One wave per 32x32 output macro-tile; K swept in steps of 32.
// ---------------------------------------------------------------------------
__global__ void __launch_bounds__(256)
gemm_f16_kernel(const _Float16* __restrict__ X, const _Float16* __restrict__ Wh,
                const float* __restrict__ bias, _Float16* __restrict__ Y,
                float* __restrict__ Yf, int Kp, int O, int Op,
                int tilesM, int tilesN) {
  int wid = blockIdx.x * (blockDim.x >> 5) + (threadIdx.x >> 5);
  if (wid >= tilesM * tilesN) return;           // whole-wave exit: EXEC all-1s
  int tm = wid % tilesM;
  int tn = wid / tilesM;
  int lane  = threadIdx.x & 31;
  int rsel  = lane & 15;
  int khalf = (lane >> 4) << 3;

  const _Float16* xa0 = X  + (size_t)(tm * 32 + rsel) * Kp + khalf;
  const _Float16* xa1 = xa0 + (size_t)16 * Kp;
  const _Float16* wb0 = Wh + (size_t)(tn * 32 + rsel) * Kp + khalf;
  const _Float16* wb1 = wb0 + (size_t)16 * Kp;

  v8f a00 = {}, a01 = {}, a10 = {}, a11 = {};
  for (int kb = 0; kb < Kp; kb += 32) {
    v16h a0 = ldfrag(xa0 + kb);
    v16h a1 = ldfrag(xa1 + kb);
    v16h b0 = ldfrag(wb0 + kb);
    v16h b1 = ldfrag(wb1 + kb);
    a00 = __builtin_amdgcn_wmma_f32_16x16x32_f16(false, a0, false, b0, (short)0, a00, false, false);
    a01 = __builtin_amdgcn_wmma_f32_16x16x32_f16(false, a0, false, b1, (short)0, a01, false, false);
    a10 = __builtin_amdgcn_wmma_f32_16x16x32_f16(false, a1, false, b0, (short)0, a10, false, false);
    a11 = __builtin_amdgcn_wmma_f32_16x16x32_f16(false, a1, false, b1, (short)0, a11, false, false);
  }
  int mB = tm * 32, nB = tn * 32;
  store_tile(a00, Y, Yf, bias, mB,      nB,      O, Op, lane);
  store_tile(a01, Y, Yf, bias, mB,      nB + 16, O, Op, lane);
  store_tile(a10, Y, Yf, bias, mB + 16, nB,      O, Op, lane);
  store_tile(a11, Y, Yf, bias, mB + 16, nB + 16, O, Op, lane);
}

// Convert f32 weights (O x K) to zero-padded f16 (Op x Kp).
__global__ void wcvt_kernel(const float* __restrict__ W, _Float16* __restrict__ Wh,
                            int O, int K, int Kp, int total) {
  for (int e = blockIdx.x * blockDim.x + threadIdx.x; e < total;
       e += gridDim.x * blockDim.x) {
    int o = e / Kp, k = e - o * Kp;
    float v = (o < O && k < K) ? W[(size_t)o * K + k] : 0.0f;
    Wh[e] = (_Float16)v;
  }
}

// ---------------------------------------------------------------------------
// Farthest point sampling (serial argmax rounds; one big block)
// ---------------------------------------------------------------------------
__global__ void __launch_bounds__(1024)
fps_kernel(const float* __restrict__ xyz, int N, int npoint,
           int* __restrict__ outIdx, float* __restrict__ dist) {
  __shared__ float sv[1024];
  __shared__ int   si[1024];
  __shared__ int   sLast;
  int t = threadIdx.x;
  for (int i = t; i < N; i += 1024) dist[i] = 1e10f;
  if (t == 0) { outIdx[0] = 0; sLast = 0; }
  __syncthreads();
  for (int it = 1; it < npoint; ++it) {
    int last = sLast;
    float px = xyz[last * 3 + 0], py = xyz[last * 3 + 1], pz = xyz[last * 3 + 2];
    float best = -1.0f; int bi = 0x7fffffff;
    for (int i = t; i < N; i += 1024) {
      float dx = xyz[i * 3 + 0] - px;
      float dy = xyz[i * 3 + 1] - py;
      float dz = xyz[i * 3 + 2] - pz;
      float d  = dx * dx + dy * dy + dz * dz;
      float o  = dist[i];
      d = d < o ? d : o;
      dist[i] = d;
      if (d > best) { best = d; bi = i; }          // first-max tie-break
    }
    sv[t] = best; si[t] = bi;
    __syncthreads();
    for (int s = 512; s > 0; s >>= 1) {
      if (t < s) {
        float ov = sv[t + s]; int oi = si[t + s];
        if (ov > sv[t] || (ov == sv[t] && oi < si[t])) { sv[t] = ov; si[t] = oi; }
      }
      __syncthreads();
    }
    if (t == 0) { outIdx[it] = si[0]; sLast = si[0]; }
    __syncthreads();
  }
}

__global__ void gather_xyz_kernel(const float* __restrict__ xyz,
                                  const int* __restrict__ idx, int np,
                                  float* __restrict__ out) {
  int e = blockIdx.x * blockDim.x + threadIdx.x;
  if (e < np * 3) {
    int p = e / 3, j = e - p * 3;
    out[e] = xyz[idx[p] * 3 + j];
  }
}

// Ball query: first ns in-radius indices (ascending); pad with first found (or 0).
__global__ void ballq_kernel(const float* __restrict__ xyz, int N,
                             const float* __restrict__ nxyz, int np,
                             float r2, int ns, int* __restrict__ idx) {
  for (int p = blockIdx.x * blockDim.x + threadIdx.x; p < np;
       p += gridDim.x * blockDim.x) {
    float cx = nxyz[p * 3 + 0], cy = nxyz[p * 3 + 1], cz = nxyz[p * 3 + 2];
    int* row = idx + (size_t)p * ns;
    int cnt = 0;
    for (int i = 0; i < N && cnt < ns; ++i) {
      float dx = xyz[i * 3 + 0] - cx;
      float dy = xyz[i * 3 + 1] - cy;
      float dz = xyz[i * 3 + 2] - cz;
      if (dx * dx + dy * dy + dz * dz < r2) row[cnt++] = i;
    }
    int fill = (cnt > 0) ? row[0] : 0;
    for (int j = cnt; j < ns; ++j) row[j] = fill;
  }
}

// Build grouped GEMM input X (M x Kp f16): [xyz - center | gathered feats | 0 pad]
__global__ void group_kernel(const float* __restrict__ xyz,
                             const float* __restrict__ feats, int Cf,
                             const float* __restrict__ nxyz,
                             const int* __restrict__ bidx, int ns,
                             _Float16* __restrict__ X, int Kp, long total) {
  long stride = (long)gridDim.x * blockDim.x;
  for (long e = blockIdx.x * blockDim.x + threadIdx.x; e < total; e += stride) {
    int  c   = (int)(e % Kp);
    long row = e / Kp;
    int  p   = (int)(row / ns);
    int  s   = (int)(row - (long)p * ns);
    int  gi  = bidx[(size_t)p * ns + s];
    float v = 0.0f;
    if (c < 3)            v = xyz[gi * 3 + c] - nxyz[p * 3 + c];
    else if (c < 3 + Cf)  v = feats[(size_t)gi * Cf + (c - 3)];
    X[e] = (_Float16)v;
  }
}

// Max over samples -> f32 feature buffer at channel offset (multi-scale concat).
__global__ void maxpool_kernel(const _Float16* __restrict__ Y, int np, int ns,
                               int O, int Op, float* __restrict__ feats,
                               int Ctot, int chOff) {
  int total = np * O;
  for (int e = blockIdx.x * blockDim.x + threadIdx.x; e < total;
       e += gridDim.x * blockDim.x) {
    int p = e / O, c = e - p * O;
    const _Float16* base = Y + (size_t)p * ns * Op + c;
    float m = -1e30f;
    for (int s = 0; s < ns; ++s) {
      float v = (float)base[(size_t)s * Op];
      m = v > m ? v : m;
    }
    feats[(size_t)p * Ctot + chOff + c] = m;
  }
}

// 3-NN + inverse-squared-distance weights (normalized), stable on ties.
__global__ void threenn_kernel(const float* __restrict__ uxyz, int Nu,
                               const float* __restrict__ kxyz, int Nk,
                               int* __restrict__ nnIdx, float* __restrict__ nnW) {
  for (int n = blockIdx.x * blockDim.x + threadIdx.x; n < Nu;
       n += gridDim.x * blockDim.x) {
    float ux = uxyz[n * 3 + 0], uy = uxyz[n * 3 + 1], uz = uxyz[n * 3 + 2];
    float d0 = 1e30f, d1 = 1e30f, d2 = 1e30f;
    int   i0 = 0, i1 = 0, i2 = 0;
    for (int k = 0; k < Nk; ++k) {
      float dx = kxyz[k * 3 + 0] - ux;
      float dy = kxyz[k * 3 + 1] - uy;
      float dz = kxyz[k * 3 + 2] - uz;
      float d  = dx * dx + dy * dy + dz * dz;
      if (d < d0)      { d2 = d1; i2 = i1; d1 = d0; i1 = i0; d0 = d; i0 = k; }
      else if (d < d1) { d2 = d1; i2 = i1; d1 = d;  i1 = k; }
      else if (d < d2) { d2 = d;  i2 = k; }
    }
    float w0 = 1.0f / (d0 + 1e-8f);
    float w1 = 1.0f / (d1 + 1e-8f);
    float w2 = 1.0f / (d2 + 1e-8f);
    float s  = w0 + w1 + w2;
    nnIdx[n * 3 + 0] = i0; nnIdx[n * 3 + 1] = i1; nnIdx[n * 3 + 2] = i2;
    nnW[n * 3 + 0] = w0 / s; nnW[n * 3 + 1] = w1 / s; nnW[n * 3 + 2] = w2 / s;
  }
}

// FP input: [interpolated known feats | skip feats | 0 pad] -> f16 X (Nu x Kp)
__global__ void interp_kernel(const int* __restrict__ nnIdx,
                              const float* __restrict__ nnW,
                              const float* __restrict__ kf, int Ck,
                              const float* __restrict__ skip, int Cs,
                              _Float16* __restrict__ X, int Kp, long total) {
  long stride = (long)gridDim.x * blockDim.x;
  for (long e = blockIdx.x * blockDim.x + threadIdx.x; e < total; e += stride) {
    int c = (int)(e % Kp);
    int n = (int)(e / Kp);
    float v = 0.0f;
    if (c < Ck) {
      v = nnW[n * 3 + 0] * kf[(size_t)nnIdx[n * 3 + 0] * Ck + c]
        + nnW[n * 3 + 1] * kf[(size_t)nnIdx[n * 3 + 1] * Ck + c]
        + nnW[n * 3 + 2] * kf[(size_t)nnIdx[n * 3 + 2] * Ck + c];
    } else if (skip != nullptr && c < Ck + Cs) {
      v = skip[(size_t)n * Cs + (c - Ck)];
    }
    X[e] = (_Float16)v;
  }
}

// Head layer 2 (64 -> 5, no ReLU), writes transposed output (5, N).
__global__ void head2_kernel(const _Float16* __restrict__ h,
                             const float* __restrict__ W,
                             const float* __restrict__ b,
                             float* __restrict__ out, int N) {
  int total = N * 5;
  for (int e = blockIdx.x * blockDim.x + threadIdx.x; e < total;
       e += gridDim.x * blockDim.x) {
    int n = e / 5, o = e - n * 5;
    float s = b[o];
    const _Float16* hr = h + (size_t)n * 64;
    const float*    wr = W + o * 64;
    for (int c = 0; c < 64; ++c) s += (float)hr[c] * wr[c];
    out[(size_t)o * N + n] = s;
  }
}

// ---------------------------------------------------------------------------
// Host side
// ---------------------------------------------------------------------------
static inline int ru32(int x) { return (x + 31) & ~31; }
static inline int gcfg(long total) {
  long b = (total + 255) / 256;
  if (b < 1) b = 1;
  if (b > 32768) b = 32768;
  return (int)b;
}

static void run_gemm(hipStream_t stream, const _Float16* X, const float* W,
                     const float* bias, _Float16* Wh, _Float16* Y, float* Yf,
                     int M, int K, int Kp, int O, int Op) {
  int wtotal = Op * Kp;
  wcvt_kernel<<<gcfg(wtotal), 256, 0, stream>>>(W, Wh, O, K, Kp, wtotal);
  int tilesM = M / 32, tilesN = Op / 32;
  int waves  = tilesM * tilesN;
  gemm_f16_kernel<<<(waves + 7) / 8, 256, 0, stream>>>(X, Wh, bias, Y, Yf,
                                                       Kp, O, Op, tilesM, tilesN);
}

extern "C" void kernel_launch(void* const* d_in, const int* in_sizes, int n_in,
                              void* d_out, int out_size, void* d_ws, size_t ws_size,
                              hipStream_t stream) {
  (void)in_sizes; (void)n_in; (void)out_size; (void)ws_size;

  // ---- unpack inputs (setup_inputs insertion order, pytree DFS) ----
  int ii = 0;
  const float* pc = (const float*)d_in[ii++];                 // (1,16384,3)
  const float *saW[4][2][3], *saB[4][2][3];
  for (int l = 0; l < 4; ++l)
    for (int s = 0; s < 2; ++s)
      for (int j = 0; j < 3; ++j) {
        saW[l][s][j] = (const float*)d_in[ii++];
        saB[l][s][j] = (const float*)d_in[ii++];
      }
  const float *fpW[4][2], *fpB[4][2];
  for (int m = 0; m < 4; ++m)
    for (int j = 0; j < 2; ++j) {
      fpW[m][j] = (const float*)d_in[ii++];
      fpB[m][j] = (const float*)d_in[ii++];
    }
  const float* hW1 = (const float*)d_in[ii++];
  const float* hB1 = (const float*)d_in[ii++];
  const float* hW2 = (const float*)d_in[ii++];
  const float* hB2 = (const float*)d_in[ii++];

  // ---- workspace carving ----
  char* wp = (char*)d_ws;
  auto alloc = [&](size_t bytes) -> char* {
    char* r = wp;
    wp += (bytes + 255) & ~(size_t)255;
    return r;
  };
  static const int NPOINT[4] = {8192, 2048, 512, 128};
  static const int CTOT[4]   = {96, 256, 512, 1024};
  float* fpsDist = (float*)alloc(16384 * 4);
  int*   fpsIdx  = (int*)alloc(8192 * 4);
  float* nxyz[4];
  for (int l = 0; l < 4; ++l) nxyz[l] = (float*)alloc((size_t)NPOINT[l] * 3 * 4);
  int* ballIdx = (int*)alloc((size_t)8192 * 32 * 4);
  _Float16* actA = (_Float16*)alloc((size_t)262144 * 64 * 2);
  _Float16* actB = (_Float16*)alloc((size_t)262144 * 64 * 2);
  _Float16* Wh   = (_Float16*)alloc((size_t)512 * 1536 * 2);
  float* feats[4];
  for (int l = 0; l < 4; ++l)
    feats[l] = (float*)alloc((size_t)NPOINT[l] * CTOT[l] * 4);
  float* fpo0 = (float*)alloc((size_t)512 * 512 * 4);
  float* fpo1 = (float*)alloc((size_t)2048 * 512 * 4);
  float* fpo2 = (float*)alloc((size_t)8192 * 256 * 4);
  int*   nnIdx = (int*)alloc((size_t)16384 * 3 * 4);
  float* nnW   = (float*)alloc((size_t)16384 * 3 * 4);

  static const float RAD[4][2] = {{0.1f, 0.5f}, {0.5f, 1.0f}, {1.0f, 2.0f}, {2.0f, 4.0f}};
  static const int   NS[4][2]  = {{16, 32}, {16, 32}, {16, 32}, {16, 32}};
  static const int   DIMS[4][2][4] = {
      {{3, 16, 16, 32},    {3, 32, 32, 64}},
      {{99, 64, 64, 128},  {99, 64, 96, 128}},
      {{259, 128, 196, 256}, {259, 128, 196, 256}},
      {{515, 256, 256, 512}, {515, 256, 384, 512}}};

  // ---- Set Abstraction stages ----
  const float* prevXyz   = pc;
  int          prevN     = 16384;
  const float* prevFeats = nullptr;

  for (int l = 0; l < 4; ++l) {
    int np = NPOINT[l];
    fps_kernel<<<1, 1024, 0, stream>>>(prevXyz, prevN, np, fpsIdx, fpsDist);
    gather_xyz_kernel<<<(np * 3 + 255) / 256, 256, 0, stream>>>(prevXyz, fpsIdx, np, nxyz[l]);
    int chOff = 0;
    for (int s = 0; s < 2; ++s) {
      float r  = RAD[l][s];
      int   ns = NS[l][s];
      ballq_kernel<<<(np + 255) / 256, 256, 0, stream>>>(prevXyz, prevN, nxyz[l],
                                                         np, r * r, ns, ballIdx);
      const int* dims = DIMS[l][s];
      int Cin = dims[0];
      int Cf  = (l == 0) ? 0 : (Cin - 3);
      int M   = np * ns;                                 // always multiple of 32
      int Kp  = ru32(Cin);
      long total = (long)M * Kp;
      group_kernel<<<gcfg(total), 256, 0, stream>>>(prevXyz, prevFeats, Cf,
                                                    nxyz[l], ballIdx, ns, actA,
                                                    Kp, total);
      _Float16* X = actA;
      _Float16* Y = actB;
      int K = Cin;
      for (int j = 0; j < 3; ++j) {
        int O  = dims[j + 1];
        int Op = ru32(O);
        run_gemm(stream, X, saW[l][s][j], saB[l][s][j], Wh, Y, nullptr, M, K, Kp, O, Op);
        K = O; Kp = Op;
        _Float16* t = X; X = Y; Y = t;                   // output now in X
      }
      int O = dims[3], Op = ru32(O);
      maxpool_kernel<<<gcfg((long)np * O), 256, 0, stream>>>(X, np, ns, O, Op,
                                                             feats[l], CTOT[l], chOff);
      chOff += O;
    }
    prevXyz   = nxyz[l];
    prevN     = np;
    prevFeats = feats[l];
  }

  // ---- Feature Propagation stages ----
  static const int FPD[4][3] = {{1536, 512, 512}, {768, 512, 512},
                                {608, 256, 256},  {256, 128, 128}};
  const float* uX[4]   = {nxyz[2], nxyz[1], nxyz[0], pc};
  int          Nu[4]   = {512, 2048, 8192, 16384};
  const float* kX[4]   = {nxyz[3], nxyz[2], nxyz[1], nxyz[0]};
  int          Nk[4]   = {128, 512, 2048, 8192};
  const float* kF[4]   = {feats[3], fpo0, fpo1, fpo2};
  int          Ck[4]   = {1024, 512, 512, 256};
  const float* skF[4]  = {feats[2], feats[1], feats[0], nullptr};
  int          Cs[4]   = {512, 256, 96, 0};
  float*       fOut[4] = {fpo0, fpo1, fpo2, nullptr};

  for (int m = 0; m < 4; ++m) {
    threenn_kernel<<<(Nu[m] + 255) / 256, 256, 0, stream>>>(uX[m], Nu[m], kX[m],
                                                            Nk[m], nnIdx, nnW);
    int  Kin   = FPD[m][0];                  // == Ck + Cs
    int  Kp    = ru32(Kin);
    long total = (long)Nu[m] * Kp;
    interp_kernel<<<gcfg(total), 256, 0, stream>>>(nnIdx, nnW, kF[m], Ck[m],
                                                   skF[m], Cs[m], actA, Kp, total);
    int O0 = FPD[m][1], Op0 = ru32(O0);
    run_gemm(stream, actA, fpW[m][0], fpB[m][0], Wh, actB, nullptr,
             Nu[m], Kin, Kp, O0, Op0);
    int O1 = FPD[m][2], Op1 = ru32(O1);
    run_gemm(stream, actB, fpW[m][1], fpB[m][1], Wh, actA, fOut[m],
             Nu[m], O0, Op0, O1, Op1);       // final act in actA
  }

  // ---- Head: 128 -> 64 (ReLU, WMMA) then 64 -> 5 (no ReLU, transposed out) ----
  run_gemm(stream, actA, hW1, hB1, Wh, actB, nullptr, 16384, 128, 128, 64, 64);
  head2_kernel<<<(16384 * 5 + 255) / 256, 256, 0, stream>>>(actB, hW2, hB2,
                                                            (float*)d_out, 16384);
}
